// MambaBlock_17703855194618
// MI455X (gfx1250) — matrix operand order repaired and checked by reading, not compiled, not run on hardware
//
#include <hip/hip_runtime.h>
#include <hip/hip_bf16.h>

// ---------------------------------------------------------------------------
// Mamba block forward for gfx1250 (MI455X).  B=2, L=2048, D=1024, N=16, K=4.
// GEMMs: v_wmma_f32_16x16x32_bf16 with all operands pre-converted to bf16 in
// global memory (weights pre-transposed to [N][K]), so the GEMM hot loop is
// pure b128 data movement + WMMA: no conversion ALU, half the DRAM bytes.
// ---------------------------------------------------------------------------

#define B_ 2
#define L_ 2048
#define D_ 1024
#define N_ 16
#define KC_ 4
#define BL_ (B_ * L_)   // 4096 rows for all GEMMs
#define KD 1024         // shared K dimension of all GEMMs
#define XPN 128         // padded xp row stride (>= 3*N_ = 48, multiple of TN)

typedef __attribute__((ext_vector_type(16))) __bf16 v16bf;
typedef __attribute__((ext_vector_type(8)))  float  v8f;

union Frag16 {
    v16bf v;
    unsigned int u[8];
    uint4 q[2];
};

__device__ __forceinline__ unsigned short f32_to_bf16_rne(float f) {
    unsigned int u = __float_as_uint(f);
    return (unsigned short)((u + 0x7FFFu + ((u >> 16) & 1u)) >> 16);
}

__device__ __forceinline__ float bf16_to_f32(unsigned short b) {
    return __uint_as_float((unsigned int)b << 16);
}

// Pack two fp32 into one dword of two bf16 (lo -> [15:0], hi -> [31:16]).
#if __has_builtin(__builtin_amdgcn_cvt_pk_bf16_f32)
__device__ __forceinline__ unsigned int pack_bf16(float lo, float hi) {
    auto r = __builtin_amdgcn_cvt_pk_bf16_f32(lo, hi);
    unsigned int u;
    __builtin_memcpy(&u, &r, sizeof(u));
    return u;
}
#else
__device__ __forceinline__ unsigned int pack_bf16(float lo, float hi) {
    unsigned int a = __float_as_uint(lo) + 0x8000u;   // round-half-up
    unsigned int b = __float_as_uint(hi) + 0x8000u;
    return __builtin_amdgcn_perm(b, a, 0x07060302u);  // splice high halves
}
#endif

__device__ __forceinline__ float silu_f(float x) {
    return x / (1.0f + __expf(-x));
}

__device__ __forceinline__ float softplus_f(float x) {
    return (x > 20.0f) ? x : log1pf(__expf(x));
}

// ---------------------------------------------------------------------------
// WMMA GEMM with bf16 operands:  C[M x N] = A[M x KD] * Bt[N x KD]^T + bias.
// A, Bt row-major bf16 with row stride KD.  M%128==0, N%128==0.
// Block tile 128x128, K-step 32, 256 threads = 8 waves (2m x 4n);
// each wave owns 64m x 32n => 4x2 accumulators => 8 v_wmma per K-step.
// Register double-buffering: next tile's global loads issue before compute.
// ---------------------------------------------------------------------------
#define TM 128
#define TN 128
#define TK 32
#define LP 40   // padded LDS row (elements): 80 B, 16B-aligned, bank-friendly

__global__ __launch_bounds__(256)
void wmma_gemm_kernel(const unsigned short* __restrict__ A,
                      const unsigned short* __restrict__ Bt,
                      const float* __restrict__ bias, float* __restrict__ C,
                      int ldc) {
    __shared__ unsigned short As[TM][LP];   // [m][k] bf16 bits (10 KB)
    __shared__ unsigned short Bs[TN][LP];   // [n][k] bf16 bits (10 KB)

    const int tid    = threadIdx.x;
    const int wave   = tid >> 5;
    const int lane   = tid & 31;
    const int h      = lane >> 4;     // lane half (WMMA layout)
    const int lane16 = lane & 15;

    const long row0 = (long)blockIdx.y * TM;
    const long col0 = (long)blockIdx.x * TN;
    const int  wm   = (wave >> 2) * 64;   // wave m-offset (2 rows of waves)
    const int  wn   = (wave & 3) * 32;    // wave n-offset (4 cols of waves)

    // staging: each thread moves 16 bf16 (32 B) of one row of A and of Bt
    const int st_row = tid >> 1;              // 0..127
    const int st_k   = (tid & 1) * 16;        // 0 or 16

    const unsigned short* aP = A  + (row0 + st_row) * KD + st_k;
    const unsigned short* bP = Bt + (col0 + st_row) * KD + st_k;

    uint4 ra0 = ((const uint4*)aP)[0], ra1 = ((const uint4*)aP)[1];
    uint4 rb0 = ((const uint4*)bP)[0], rb1 = ((const uint4*)bP)[1];

    v8f acc[4][2] = {};

    for (int kt = 0; kt < KD; kt += TK) {
        __syncthreads();   // previous compute done; LDS reusable

        *(uint4*)&As[st_row][st_k]     = ra0;
        *(uint4*)&As[st_row][st_k + 8] = ra1;
        *(uint4*)&Bs[st_row][st_k]     = rb0;
        *(uint4*)&Bs[st_row][st_k + 8] = rb1;
        __syncthreads();   // LDS tile ready

        // issue next tile's global loads; latency hidden by WMMAs
        if (kt + TK < KD) {
            const unsigned short* aN = aP + kt + TK;
            const unsigned short* bN = bP + kt + TK;
            ra0 = ((const uint4*)aN)[0]; ra1 = ((const uint4*)aN)[1];
            rb0 = ((const uint4*)bN)[0]; rb1 = ((const uint4*)bN)[1];
        }

        // ---- fragments (ISA 7.12.2 16-bit layouts), all ds_load_b128 ----
        // A: lane(h,lane16) row m: u[0..3] = K h*8+{0..7}, u[4..7] = K 16+h*8+{0..7}
        Frag16 fa[4];
        #pragma unroll
        for (int i = 0; i < 4; ++i) {
            const int am = wm + i * 16 + lane16;
            fa[i].q[0] = *(const uint4*)&As[am][h * 8];
            fa[i].q[1] = *(const uint4*)&As[am][16 + h * 8];
        }
        // B: lane(h,lane16) col n: u[0..3] = K h*8+{0..7}, u[4..7] = K 16+h*8+{0..7}
        Frag16 fb[2];
        #pragma unroll
        for (int s = 0; s < 2; ++s) {
            const int n = wn + s * 16 + lane16;
            fb[s].q[0] = *(const uint4*)&Bs[n][h * 8];
            fb[s].q[1] = *(const uint4*)&Bs[n][16 + h * 8];
        }

        #pragma unroll
        for (int i = 0; i < 4; ++i)
            #pragma unroll
            for (int s = 0; s < 2; ++s)
                acc[i][s] = __builtin_amdgcn_wmma_f32_16x16x32_bf16(
                    false, fa[i].v, false, fb[s].v, (short)0, acc[i][s],
                    false, false);
    }

    // ---- epilogue: C/D layout: VGPR r -> M = (lane/16)*8 + r, N = lane%16 ----
    #pragma unroll
    for (int s = 0; s < 2; ++s) {
        const long n   = col0 + wn + s * 16 + lane16;
        const float bv = bias ? bias[n] : 0.0f;
        #pragma unroll
        for (int i = 0; i < 4; ++i) {
            const long mbase = row0 + wm + i * 16 + h * 8;
            #pragma unroll
            for (int r = 0; r < 8; ++r)
                C[(mbase + r) * ldc + n] = acc[i][s][r] + bv;
        }
    }
}

// ---------------------------------------------------------------------------
// LDS-tiled convert+transpose: dst[c][r] (bf16) = src[r][c] (fp32), with
// zero-fill for c >= Cv.  R, C multiples of 32.  Block (32,8), 32x32 tiles.
// ---------------------------------------------------------------------------
__global__ __launch_bounds__(256)
void transpose_cvt_kernel(const float* __restrict__ src,
                          unsigned short* __restrict__ dst,
                          int R, int Cv, int ldsrc) {
    __shared__ float t[32][33];
    const int c0 = blockIdx.x * 32;
    const int r0 = blockIdx.y * 32;
    #pragma unroll
    for (int i = 0; i < 4; ++i) {
        int r = r0 + threadIdx.y + i * 8;
        int c = c0 + threadIdx.x;
        t[threadIdx.y + i * 8][threadIdx.x] =
            (c < Cv) ? src[(long)r * ldsrc + c] : 0.0f;
    }
    __syncthreads();
    #pragma unroll
    for (int i = 0; i < 4; ++i) {
        int c = c0 + threadIdx.y + i * 8;   // output row = source column
        int r = r0 + threadIdx.x;           // output col = source row
        dst[(long)c * R + r] = f32_to_bf16_rne(t[threadIdx.x][threadIdx.y + i * 8]);
    }
}

// ---------------------------------------------------------------------------
// fp32 -> bf16 flat convert (vectorized: 4 floats -> 1 dword pair store)
// ---------------------------------------------------------------------------
__global__ __launch_bounds__(256)
void cvt_bf16_kernel(const float* __restrict__ src, unsigned short* __restrict__ dst,
                     int n4) {
    int i = blockIdx.x * 256 + threadIdx.x;    // over n/4
    if (i >= n4) return;
    float4 v = ((const float4*)src)[i];
    uint2 p;
    p.x = pack_bf16(v.x, v.y);
    p.y = pack_bf16(v.z, v.w);
    ((uint2*)dst)[i] = p;
}

// ---------------------------------------------------------------------------
// Depthwise causal conv (K=4) + SiLU over the xs half of proj (row stride 2D);
// writes u as bf16.
// ---------------------------------------------------------------------------
__global__ __launch_bounds__(256)
void conv_silu_kernel(const float* __restrict__ proj, const float* __restrict__ wconv,
                      const float* __restrict__ bconv, unsigned short* __restrict__ u) {
    int idx = blockIdx.x * 256 + threadIdx.x;       // over BL_*D_
    if (idx >= BL_ * D_) return;
    int d  = idx & (D_ - 1);
    int bl = idx >> 10;                              // / D_
    int b  = bl >> 11;                               // / L_
    int l  = bl & (L_ - 1);

    float acc = bconv[d];
    #pragma unroll
    for (int j = 0; j < KC_; ++j) {
        int ll = l - (KC_ - 1) + j;
        if (ll >= 0) {
            float xv = proj[((long)(b * L_ + ll)) * (2 * D_) + d];
            acc = fmaf(xv, wconv[d * KC_ + j], acc);
        }
    }
    u[idx] = f32_to_bf16_rne(silu_f(acc));
}

// ---------------------------------------------------------------------------
// delta[b,l,d] = softplus( sum_n xp[bl, n] * w_dt[n,d] + b_dt[d] )
// xp row stride = XPN;  [0:16) = delta_raw
// ---------------------------------------------------------------------------
__global__ __launch_bounds__(256)
void delta_kernel(const float* __restrict__ xp, const float* __restrict__ w_dt,
                  const float* __restrict__ b_dt, float* __restrict__ delta) {
    int idx = blockIdx.x * 256 + threadIdx.x;
    if (idx >= BL_ * D_) return;
    int d  = idx & (D_ - 1);
    int bl = idx >> 10;

    const float* xr = &xp[(long)bl * XPN];
    float acc = b_dt[d];
    #pragma unroll
    for (int n = 0; n < N_; ++n)
        acc = fmaf(xr[n], w_dt[n * D_ + d], acc);
    delta[idx] = softplus_f(acc);
}

// ---------------------------------------------------------------------------
// Selective scan, serial over L. One lane per (b,d); 16-entry state in VGPRs.
// B_t / C_t broadcast via LDS.  Fuses  + u*D_param  and  + silu(res).
// Reads u as bf16, writes y2 as bf16 (operand of the final GEMM).
// ---------------------------------------------------------------------------
__global__ __launch_bounds__(256)
void scan_kernel(const float* __restrict__ delta, const unsigned short* __restrict__ u,
                 const float* __restrict__ xp, const float* __restrict__ proj,
                 const float* __restrict__ A_log, const float* __restrict__ D_param,
                 unsigned short* __restrict__ y2) {
    __shared__ float BC[2 * N_];                    // B_t then C_t

    int g = blockIdx.x * 256 + threadIdx.x;         // 0..2047 over (b,d)
    int d = g & (D_ - 1);
    int b = g >> 10;                                 // blocks 0-3 -> b=0, 4-7 -> b=1
    int tid = threadIdx.x;

    float a[N_];
    float s[N_];
    #pragma unroll
    for (int n = 0; n < N_; ++n) {
        a[n] = -__expf(A_log[d * N_ + n]);
        s[n] = 0.0f;
    }
    const float Dp = D_param[d];

    for (int l = 0; l < L_; ++l) {
        long bl = (long)b * L_ + l;
        if (tid < 2 * N_) {
            BC[tid] = xp[bl * XPN + N_ + tid];       // [16:32)=B, [32:48)=C
        }
        __syncthreads();

        float dt = delta[bl * D_ + d];
        float ut = bf16_to_f32(u[bl * D_ + d]);
        float y = 0.0f;
        #pragma unroll
        for (int n = 0; n < N_; ++n) {
            s[n] = fmaf(dt * a[n], s[n], dt * BC[n] * ut);
            y = fmaf(s[n], BC[N_ + n], y);
        }
        float r = proj[bl * (2 * D_) + D_ + d];     // res half
        y = fmaf(ut, Dp, y) + silu_f(r);
        y2[bl * D_ + d] = f32_to_bf16_rne(y);
        __syncthreads();
    }
}

// ---------------------------------------------------------------------------
// Host-side launch
// ---------------------------------------------------------------------------
extern "C" void kernel_launch(void* const* d_in, const int* in_sizes, int n_in,
                              void* d_out, int out_size, void* d_ws, size_t ws_size,
                              hipStream_t stream) {
    const float* x       = (const float*)d_in[0];
    const float* w_in    = (const float*)d_in[1];
    const float* b_in    = (const float*)d_in[2];
    const float* w_conv  = (const float*)d_in[3];
    const float* b_conv  = (const float*)d_in[4];
    const float* w_xp    = (const float*)d_in[5];
    const float* w_dt    = (const float*)d_in[6];
    const float* b_dt    = (const float*)d_in[7];
    const float* w_out   = (const float*)d_in[8];
    const float* b_out   = (const float*)d_in[9];
    const float* A_log   = (const float*)d_in[10];
    const float* D_param = (const float*)d_in[11];
    float* out = (float*)d_out;

    // workspace layout (mixed fp32 / bf16)
    char* wsb = (char*)d_ws;
    float* proj            = (float*)wsb;                       // 4096x2048 f32 (32 MB)
    wsb += (size_t)BL_ * 2 * D_ * sizeof(float);
    float* xp              = (float*)wsb;                       // 4096x128  f32 (2 MB)
    wsb += (size_t)BL_ * XPN * sizeof(float);
    float* delta           = (float*)wsb;                       // 4096x1024 f32 (16 MB)
    wsb += (size_t)BL_ * D_ * sizeof(float);
    unsigned short* x_bf   = (unsigned short*)wsb;              // 4096x1024 bf16
    wsb += (size_t)BL_ * D_ * sizeof(unsigned short);
    unsigned short* u_bf   = (unsigned short*)wsb;              // 4096x1024 bf16
    wsb += (size_t)BL_ * D_ * sizeof(unsigned short);
    unsigned short* y2_bf  = (unsigned short*)wsb;              // 4096x1024 bf16
    wsb += (size_t)BL_ * D_ * sizeof(unsigned short);
    unsigned short* winT   = (unsigned short*)wsb;              // 2048x1024 bf16
    wsb += (size_t)(2 * D_) * KD * sizeof(unsigned short);
    unsigned short* wxpT   = (unsigned short*)wsb;              // 128x1024  bf16
    wsb += (size_t)XPN * KD * sizeof(unsigned short);
    unsigned short* woutT  = (unsigned short*)wsb;              // 1024x1024 bf16

    const int elems   = BL_ * D_;
    const int eblocks = (elems + 255) / 256;

    // --- one-shot operand preparation (bf16, weights transposed to [N][K]) ---
    cvt_bf16_kernel<<<(elems / 4 + 255) / 256, 256, 0, stream>>>(x, x_bf, elems / 4);
    {   // w_in (1024 x 2048) -> winT (2048 x 1024)
        dim3 grid((2 * D_) / 32, KD / 32);
        transpose_cvt_kernel<<<grid, dim3(32, 8), 0, stream>>>(w_in, winT, KD, 2 * D_, 2 * D_);
    }
    {   // w_xp (1024 x 48) -> wxpT (128 x 1024), zero-padded columns
        dim3 grid(XPN / 32, KD / 32);
        transpose_cvt_kernel<<<grid, dim3(32, 8), 0, stream>>>(w_xp, wxpT, KD, 3 * N_, 3 * N_);
    }
    {   // w_out (1024 x 1024) -> woutT (1024 x 1024)
        dim3 grid(D_ / 32, KD / 32);
        transpose_cvt_kernel<<<grid, dim3(32, 8), 0, stream>>>(w_out, woutT, KD, D_, D_);
    }

    // 1) proj = x @ w_in + b_in   (M=4096, N=2048)
    {
        dim3 grid((2 * D_) / TN, BL_ / TM);
        wmma_gemm_kernel<<<grid, 256, 0, stream>>>(x_bf, winT, b_in, proj, 2 * D_);
    }
    // 2) conv + silu -> u (bf16)
    conv_silu_kernel<<<eblocks, 256, 0, stream>>>(proj, w_conv, b_conv, u_bf);
    // 3) xp = u @ w_xp   (M=4096, N=128 padded)
    {
        dim3 grid(XPN / TN, BL_ / TM);
        wmma_gemm_kernel<<<grid, 256, 0, stream>>>(u_bf, wxpT, nullptr, xp, XPN);
    }
    // 4) delta = softplus(delta_raw @ w_dt + b_dt)
    delta_kernel<<<eblocks, 256, 0, stream>>>(xp, w_dt, b_dt, delta);
    // 5) selective scan (+ u*D + silu(res)) -> y2 (bf16)
    scan_kernel<<<(B_ * D_) / 256, 256, 0, stream>>>(
        delta, u_bf, xp, proj, A_log, D_param, y2_bf);
    // 6) out = y2 @ w_out + b_out   (M=4096, N=1024)
    {
        dim3 grid(D_ / TN, BL_ / TM);
        wmma_gemm_kernel<<<grid, 256, 0, stream>>>(y2_bf, woutT, b_out, out, D_);
    }
}
